// HawkBlock_87076166960154
// MI455X (gfx1250) — matrix operand order repaired and checked by reading, not compiled, not run on hardware
//
#include <hip/hip_runtime.h>
#include <hip/hip_bf16.h>

// ---------------------------------------------------------------------------
// HawkBlock on gfx1250: bf16 WMMA GEMMs (v_wmma_f32_16x16x32_bf16) with TDM
// (tensor_load_to_lds) double-buffered LDS staging for ALL GEMMs, TDM pad
// mode for bank-conflict-free ds_load_b128, fused epilogues, sequential
// hawk scan with global prefetch.
// ---------------------------------------------------------------------------

typedef __bf16 bf16_t;
typedef __bf16 v16bf __attribute__((ext_vector_type(16)));
typedef __bf16 v8bf  __attribute__((ext_vector_type(8)));
typedef float  v8f   __attribute__((ext_vector_type(8)));
typedef float  v4f   __attribute__((ext_vector_type(4)));

typedef unsigned int u32x4 __attribute__((ext_vector_type(4)));
typedef int          i32x4 __attribute__((ext_vector_type(4)));
typedef int          i32x8 __attribute__((ext_vector_type(8)));

#define HIDDEN 1024
#define D_REC  2048
#define INTER  4096
#define NB     4
#define NT     4096
#define MTOK   (NB * NT)   // 16384 tokens

// Single-B GEMM: staged K-tile 64, padded LDS row stride 72 halfs (144B).
#define KT1  64
#define STR1 72
// Dual-B GEMM: staged K-tile 32, padded LDS row stride 40 halfs (80B).
#define KT2  32
#define STR2 40

// Toolchain detection: therock-10.0 headers (amdgpu-toolchain, clang-23) ship
// the TDM header and use the 6-arg builtin; ROCm 7.2 (clang-22) uses 5 args.
#if defined(__has_include)
#if __has_include(<hip/amd_detail/amd_gfx1250_TDM.h>)
#define TDM_6ARG 1
#else
#define TDM_6ARG 0
#endif
#else
#define TDM_6ARG 0
#endif

#if defined(__has_builtin)
#if __has_builtin(__builtin_amdgcn_tensor_load_to_lds) && \
    __has_builtin(__builtin_amdgcn_s_wait_tensorcnt)
#define USE_TDM 1
#else
#define USE_TDM 0
#endif
#else
#define USE_TDM 0
#endif

__device__ __forceinline__ float sigmoidf_(float x) {
    return 1.0f / (1.0f + __expf(-x));
}

// ---------------------------------------------------------------------------
// WMMA fragment loaders (lda in elements; works for global or LDS tiles —
// addrspace inference emits ds_load_b128 for the LDS case).
//
// 16-bit A fragment 16x32 (MxK), wave32 (ISA 7.12.2):
//   lane m (0..15):  VGPR0-3 = K[0..7],  VGPR4-7 = K[16..23]   of row m
//   lane m+16:       VGPR0-3 = K[8..15], VGPR4-7 = K[24..31]   of row m
// 16-bit B fragment 32x16 (KxN):
//   lane n (0..15):  VGPR0-3 = K[0..7],  VGPR4-7 = K[16..23]  of column n  (= row n of W)
//   lane n+16:       VGPR0-3 = K[8..15], VGPR4-7 = K[24..31]  of column n
// ---------------------------------------------------------------------------
__device__ __forceinline__ v16bf load_a_frag(const bf16_t* A, int lda,
                                             int m0, int k0, int lane) {
    const int m  = m0 + (lane & 15);
    const int hi = (lane >> 4) << 3;           // 0 or 8
    const bf16_t* p = A + (size_t)m * lda + (k0 + hi);
    v8bf lo = *(const v8bf*)(p);               // K = k0+hi    .. +7
    v8bf hv = *(const v8bf*)(p + 16);          // K = k0+16+hi .. +7
    v16bf r;
#pragma unroll
    for (int i = 0; i < 8; ++i) { r[i] = lo[i]; r[8 + i] = hv[i]; }
    return r;
}

__device__ __forceinline__ v16bf load_b_frag(const bf16_t* B, int ldb,
                                             int n0, int k0, int lane) {
    const int n  = n0 + (lane & 15);
    const int ko = (lane >> 4) << 3;           // 0 or 8
    const bf16_t* p = B + (size_t)n * ldb + (k0 + ko);
    v8bf lo = *(const v8bf*)(p);
    v8bf hv = *(const v8bf*)(p + 16);
    v16bf r;
#pragma unroll
    for (int i = 0; i < 8; ++i) { r[i] = lo[i]; r[8 + i] = hv[i]; }
    return r;
}

__device__ __forceinline__ v8f wmma_bf16(v16bf a, v16bf b, v8f c) {
    return __builtin_amdgcn_wmma_f32_16x16x32_bf16(
        /*neg_a=*/false, a, /*neg_b=*/false, b,
        /*c_mod=*/(short)0, c, /*reuse_a=*/false, /*reuse_b=*/false);
}

#if USE_TDM
// ---------------------------------------------------------------------------
// TDM: 2-D tile load, data_size = 2 bytes, LDS padding enabled for bank-
// conflict-free ds_loads.  D# per ISA 08_async_tensor §8:
//  group0: d0 = count=1; d1 = lds byte addr; d2 = gaddr[31:0];
//          d3 = gaddr[56:32] | type(2)<<30
//  group1: d0 = data_size(1)<<16 | pad_en<<20 | pad_int<<22 | pad_amt<<25
//          tensor_dim0/1 = 2^30 (no clipping);
//          d3[31:16] = tile_dim0 (K elems); d4[15:0] = tile_dim1 (rows);
//          d5 = tensor_dim0_stride (elems).
//  pad codes: (4,3) = +16B per 128B row  -> stride 144B (KT1)
//             (3,3) = +16B per  64B row  -> stride  80B (KT2)
// ---------------------------------------------------------------------------
__device__ __forceinline__ void tdm_load_tile(const bf16_t* gaddr,
                                              unsigned lds_byte_off,
                                              unsigned tile_k, unsigned tile_rows,
                                              unsigned stride_elems,
                                              unsigned pad_int, unsigned pad_amt) {
    unsigned long long ga = (unsigned long long)(const void*)gaddr;
    u32x4 g0;
    g0[0] = 1u;                                                 // count=1
    g0[1] = lds_byte_off;                                       // LDS addr
    g0[2] = (unsigned)ga;                                       // gaddr lo
    g0[3] = (unsigned)((ga >> 32) & 0x01FFFFFFu) | (2u << 30);  // gaddr hi|type
    i32x8 g1;
    g1[0] = (int)((1u << 16) | (1u << 20) | (pad_int << 22) | (pad_amt << 25));
    g1[1] = 0;                                  // tensor_dim0[15:0] = 0
    g1[2] = (int)0x4000u;                       // tensor_dim0 = 2^30
    g1[3] = (int)(0x4000u | (tile_k << 16));    // tensor_dim1 = 2^30 | tile_dim0
    g1[4] = (int)(tile_rows & 0xFFFFu);         // tile_dim1, tile_dim2=0
    g1[5] = (int)stride_elems;                  // tensor_dim0_stride lo
    g1[6] = 0;                                  // stride0 hi, stride1 lo
    g1[7] = 0;
    i32x4 z4 = {0, 0, 0, 0};
#if TDM_6ARG
    i32x8 z8 = {0, 0, 0, 0, 0, 0, 0, 0};
    __builtin_amdgcn_tensor_load_to_lds(g0, g1, z4, z4, z8, 0);
#else
    __builtin_amdgcn_tensor_load_to_lds(g0, g1, z4, z4, 0);
#endif
}

__device__ __forceinline__ unsigned lds_off_of(const void* p) {
    // Generic pointers to LDS carry the LDS byte offset in the low 32 bits.
    return (unsigned)(unsigned long long)p;
}
#else
// Fallback: cooperative per-thread staging into padded layout.
__device__ __forceinline__ void stage_fallback(bf16_t* dst, const bf16_t* src,
                                               int ldsrc, int rows, int kelems,
                                               int dststr, int tid) {
    const int cpr = kelems >> 3;   // 16B chunks per row
    for (int c = tid; c < rows * cpr; c += 128) {
        const int row = c / cpr;
        const int h = c - row * cpr;
        *(v8bf*)&dst[row * dststr + h * 8] =
            *(const v8bf*)(src + (size_t)row * ldsrc + h * 8);
    }
}
#endif  // USE_TDM

// ---------------------------------------------------------------------------
// Single-B GEMM: TDM double-buffered LDS staging, K-tile 64.
// Block: 128 threads = 4 waves stacked in M -> block tile 128 x 64.
// Wave tile 32 x 64 = 2x4 WMMA accumulators.
//   mode 0: out = sigmoid(bias[col] + acc)     (f32)  [w_a path]
//   mode 1: out = resid[row*N+col] + acc       (f32)  [w_out / w_down]
// ---------------------------------------------------------------------------
__global__ void __launch_bounds__(128)
gemm_single(const bf16_t* __restrict__ A, const bf16_t* __restrict__ B,
            int N, int K, float* __restrict__ outf,
            const float* __restrict__ bias, const float* __restrict__ resid,
            int mode)
{
    __shared__ bf16_t aT[2][128 * STR1];   // 2 x 18KB
    __shared__ bf16_t bT[2][64 * STR1];    // 2 x  9KB

    const int lane = threadIdx.x & 31;
    const int wave = threadIdx.x >> 5;
    const int n0   = blockIdx.x * 64;
    const int mblk = blockIdx.y * 128;
    const int m0l  = wave * 32;            // wave's row offset inside A tile

    v8f acc[2][4];
#pragma unroll
    for (int i = 0; i < 2; ++i)
#pragma unroll
        for (int j = 0; j < 4; ++j) acc[i][j] = (v8f)0.0f;

    const int nkt = K / KT1;

#if USE_TDM
    if (wave == 0) {
        tdm_load_tile(A + (size_t)mblk * K, lds_off_of(&aT[0][0]), KT1, 128,
                      (unsigned)K, 4, 3);
        tdm_load_tile(B + (size_t)n0 * K,   lds_off_of(&bT[0][0]), KT1, 64,
                      (unsigned)K, 4, 3);
    }
#endif

    for (int kt = 0; kt < nkt; ++kt) {
        const int cur = kt & 1;
#if USE_TDM
        if (wave == 0) {
            if (kt + 1 < nkt) {
                const int k0n = (kt + 1) * KT1;
                tdm_load_tile(A + (size_t)mblk * K + k0n, lds_off_of(&aT[cur ^ 1][0]),
                              KT1, 128, (unsigned)K, 4, 3);
                tdm_load_tile(B + (size_t)n0 * K + k0n, lds_off_of(&bT[cur ^ 1][0]),
                              KT1, 64, (unsigned)K, 4, 3);
                __builtin_amdgcn_s_wait_tensorcnt(2);  // current pair complete
            } else {
                __builtin_amdgcn_s_wait_tensorcnt(0);
            }
        }
        __syncthreads();
#else
        __syncthreads();
        stage_fallback(&aT[cur][0], A + (size_t)mblk * K + kt * KT1, K, 128, KT1,
                       STR1, threadIdx.x);
        stage_fallback(&bT[cur][0], B + (size_t)n0 * K + kt * KT1, K, 64, KT1,
                       STR1, threadIdx.x);
        __syncthreads();
#endif
#pragma unroll
        for (int ks = 0; ks < KT1; ks += 32) {
            v16bf af[2];
#pragma unroll
            for (int i = 0; i < 2; ++i)
                af[i] = load_a_frag(&aT[cur][0], STR1, m0l + 16 * i, ks, lane);
            v16bf bfr[4];
#pragma unroll
            for (int j = 0; j < 4; ++j)
                bfr[j] = load_b_frag(&bT[cur][0], STR1, 16 * j, ks, lane);
#pragma unroll
            for (int i = 0; i < 2; ++i)
#pragma unroll
                for (int j = 0; j < 4; ++j)
                    acc[i][j] = wmma_bf16(af[i], bfr[j], acc[i][j]);
        }
        __syncthreads();   // all waves done reading before buffers are reused
    }

    // C/D layout: VGPR r, lane l: row = m + (l>=16 ? 8 : 0) + r, col = n + (l&15)
    const int colL = lane & 15;
    const int rowL = (lane >> 4) << 3;
#pragma unroll
    for (int i = 0; i < 2; ++i) {
#pragma unroll
        for (int j = 0; j < 4; ++j) {
            const int col  = n0 + 16 * j + colL;
            const int row0 = mblk + m0l + 16 * i + rowL;
#pragma unroll
            for (int r = 0; r < 8; ++r) {
                const int row = row0 + r;
                const float v = acc[i][j][r];
                float o;
                if (mode == 0) o = sigmoidf_(bias[col] + v);
                else           o = resid[(size_t)row * N + col] + v;
                outf[(size_t)row * N + col] = o;
            }
        }
    }
}

// ---------------------------------------------------------------------------
// Dual-B GEMM: TDM staging of A, B1, B2 tiles (K-tile 32), shared A frags.
//   mode 0: out1 = bf16(acc1)             [x_in]
//           out2 = bf16(sigmoid(acc2))    [gate]
//   mode 1: out1 = bf16(silu(acc1)*acc2)  [MLP activation]
// ---------------------------------------------------------------------------
__global__ void __launch_bounds__(128)
gemm_dual(const bf16_t* __restrict__ A,
          const bf16_t* __restrict__ B1, const bf16_t* __restrict__ B2,
          int N, int K, bf16_t* __restrict__ out1, bf16_t* __restrict__ out2,
          int mode)
{
    __shared__ bf16_t aT[2][128 * STR2];   // 2 x 10KB
    __shared__ bf16_t b1T[2][64 * STR2];   // 2 x  5KB
    __shared__ bf16_t b2T[2][64 * STR2];   // 2 x  5KB

    const int lane = threadIdx.x & 31;
    const int wave = threadIdx.x >> 5;
    const int n0   = blockIdx.x * 64;
    const int mblk = blockIdx.y * 128;
    const int m0l  = wave * 32;

    v8f acc1[2][4], acc2[2][4];
#pragma unroll
    for (int i = 0; i < 2; ++i)
#pragma unroll
        for (int j = 0; j < 4; ++j) { acc1[i][j] = (v8f)0.0f; acc2[i][j] = (v8f)0.0f; }

    const int nkt = K / KT2;

#if USE_TDM
    if (wave == 0) {
        tdm_load_tile(A  + (size_t)mblk * K, lds_off_of(&aT[0][0]),  KT2, 128,
                      (unsigned)K, 3, 3);
        tdm_load_tile(B1 + (size_t)n0 * K,   lds_off_of(&b1T[0][0]), KT2, 64,
                      (unsigned)K, 3, 3);
        tdm_load_tile(B2 + (size_t)n0 * K,   lds_off_of(&b2T[0][0]), KT2, 64,
                      (unsigned)K, 3, 3);
    }
#endif

    for (int kt = 0; kt < nkt; ++kt) {
        const int cur = kt & 1;
#if USE_TDM
        if (wave == 0) {
            if (kt + 1 < nkt) {
                const int k0n = (kt + 1) * KT2;
                tdm_load_tile(A  + (size_t)mblk * K + k0n, lds_off_of(&aT[cur ^ 1][0]),
                              KT2, 128, (unsigned)K, 3, 3);
                tdm_load_tile(B1 + (size_t)n0 * K + k0n, lds_off_of(&b1T[cur ^ 1][0]),
                              KT2, 64, (unsigned)K, 3, 3);
                tdm_load_tile(B2 + (size_t)n0 * K + k0n, lds_off_of(&b2T[cur ^ 1][0]),
                              KT2, 64, (unsigned)K, 3, 3);
                __builtin_amdgcn_s_wait_tensorcnt(3);  // current triple complete
            } else {
                __builtin_amdgcn_s_wait_tensorcnt(0);
            }
        }
        __syncthreads();
#else
        __syncthreads();
        stage_fallback(&aT[cur][0],  A  + (size_t)mblk * K + kt * KT2, K, 128, KT2,
                       STR2, threadIdx.x);
        stage_fallback(&b1T[cur][0], B1 + (size_t)n0 * K + kt * KT2, K, 64, KT2,
                       STR2, threadIdx.x);
        stage_fallback(&b2T[cur][0], B2 + (size_t)n0 * K + kt * KT2, K, 64, KT2,
                       STR2, threadIdx.x);
        __syncthreads();
#endif
        v16bf af[2];
#pragma unroll
        for (int i = 0; i < 2; ++i)
            af[i] = load_a_frag(&aT[cur][0], STR2, m0l + 16 * i, 0, lane);
#pragma unroll
        for (int j = 0; j < 4; ++j) {
            v16bf b1 = load_b_frag(&b1T[cur][0], STR2, 16 * j, 0, lane);
#pragma unroll
            for (int i = 0; i < 2; ++i) acc1[i][j] = wmma_bf16(af[i], b1, acc1[i][j]);
        }
#pragma unroll
        for (int j = 0; j < 4; ++j) {
            v16bf b2 = load_b_frag(&b2T[cur][0], STR2, 16 * j, 0, lane);
#pragma unroll
            for (int i = 0; i < 2; ++i) acc2[i][j] = wmma_bf16(af[i], b2, acc2[i][j]);
        }
        __syncthreads();
    }

    const int colL = lane & 15;
    const int rowL = (lane >> 4) << 3;
#pragma unroll
    for (int i = 0; i < 2; ++i) {
#pragma unroll
        for (int j = 0; j < 4; ++j) {
            const int col  = n0 + 16 * j + colL;
            const int row0 = mblk + m0l + 16 * i + rowL;
#pragma unroll
            for (int r = 0; r < 8; ++r) {
                const int row = row0 + r;
                const size_t idx = (size_t)row * N + col;
                const float v1 = acc1[i][j][r];
                const float v2 = acc2[i][j][r];
                if (mode == 0) {
                    out1[idx] = (bf16_t)v1;
                    out2[idx] = (bf16_t)sigmoidf_(v2);
                } else {
                    out1[idx] = (bf16_t)((v1 * sigmoidf_(v1)) * v2);  // silu(g)*u
                }
            }
        }
    }
}

// ---------------------------------------------------------------------------
// RMSNorm: one block per row of 1024, writes bf16.
// ---------------------------------------------------------------------------
__global__ void __launch_bounds__(256)
rmsnorm_kernel(const float* __restrict__ x, const float* __restrict__ w,
               bf16_t* __restrict__ out)
{
    __shared__ float red[256];
    const int row = blockIdx.x;
    const int t = threadIdx.x;
    const float* xr = x + (size_t)row * HIDDEN;
    v4f v = *(const v4f*)(xr + t * 4);
    red[t] = v[0] * v[0] + v[1] * v[1] + v[2] * v[2] + v[3] * v[3];
    __syncthreads();
#pragma unroll
    for (int s = 128; s > 0; s >>= 1) {
        if (t < s) red[t] += red[t + s];
        __syncthreads();
    }
    const float inv = rsqrtf(red[0] * (1.0f / HIDDEN) + 1e-6f);
    bf16_t* o = out + (size_t)row * HIDDEN + t * 4;
    const float* wr = w + t * 4;
#pragma unroll
    for (int c = 0; c < 4; ++c) o[c] = (bf16_t)(v[c] * inv * wr[c]);
}

// ---------------------------------------------------------------------------
// Hawk scan: h_t = a_t*h_{t-1} + sqrt(1 - a_t^2 + 1e-8)*x_t ; y = gate*h.
// One thread per (b,d) channel, coalesced across d each timestep; prefetch a
// few timesteps ahead (global_prefetch) to hide HBM latency in the serial
// recurrence. Rewrites gate buffer in place.
// ---------------------------------------------------------------------------
__global__ void __launch_bounds__(256)
hawk_scan_kernel(const float* __restrict__ a, const bf16_t* __restrict__ xin,
                 bf16_t* gate_y)
{
    const int tid = blockIdx.x * blockDim.x + threadIdx.x;  // 0 .. NB*D_REC-1
    const int b = tid / D_REC;
    const int d = tid % D_REC;
    size_t idx = (size_t)b * NT * D_REC + d;
    float h = 0.0f;
    for (int t = 0; t < NT; ++t, idx += D_REC) {
        if (t + 8 < NT) {
            __builtin_prefetch(a + idx + (size_t)8 * D_REC, 0, 3);
            __builtin_prefetch(xin + idx + (size_t)8 * D_REC, 0, 3);
            __builtin_prefetch(gate_y + idx + (size_t)8 * D_REC, 0, 3);
        }
        const float at = a[idx];
        const float xt = (float)xin[idx];
        h = at * h + sqrtf(1.0f - at * at + 1e-8f) * xt;
        const float g = (float)gate_y[idx];
        gate_y[idx] = (bf16_t)(g * h);
    }
}

// ---------------------------------------------------------------------------
// f32 -> bf16 weight convert
// ---------------------------------------------------------------------------
__global__ void __launch_bounds__(256)
convert_f32_bf16(const float* __restrict__ s, bf16_t* __restrict__ d, int n)
{
    const int i = blockIdx.x * blockDim.x + threadIdx.x;
    if (i < n) d[i] = (bf16_t)s[i];
}

// ---------------------------------------------------------------------------
extern "C" void kernel_launch(void* const* d_in, const int* in_sizes, int n_in,
                              void* d_out, int out_size, void* d_ws, size_t ws_size,
                              hipStream_t stream)
{
    (void)in_sizes; (void)n_in; (void)out_size; (void)ws_size;

    const float* x       = (const float*)d_in[0];   // [MTOK, HIDDEN]
    const float* ln1_w   = (const float*)d_in[1];
    const float* ln2_w   = (const float*)d_in[2];
    const float* w_in    = (const float*)d_in[3];   // [D_REC, HIDDEN]
    const float* w_gate  = (const float*)d_in[4];   // [D_REC, HIDDEN]
    const float* a_param = (const float*)d_in[5];   // [D_REC]
    const float* w_a     = (const float*)d_in[6];   // [D_REC, D_REC]
    const float* w_out   = (const float*)d_in[7];   // [HIDDEN, D_REC]
    const float* w_mg    = (const float*)d_in[8];   // [INTER, HIDDEN]
    const float* w_mu    = (const float*)d_in[9];   // [INTER, HIDDEN]
    const float* w_md    = (const float*)d_in[10];  // [HIDDEN, INTER]

    char* ws = (char*)d_ws;
    size_t off = 0;
    auto alloc = [&](size_t bytes) -> void* {
        void* p = ws + off;
        off += (bytes + 255) & ~(size_t)255;
        return p;
    };

    bf16_t* wInb   = (bf16_t*)alloc((size_t)D_REC * HIDDEN * 2);
    bf16_t* wGateb = (bf16_t*)alloc((size_t)D_REC * HIDDEN * 2);
    bf16_t* wAb    = (bf16_t*)alloc((size_t)D_REC * D_REC * 2);
    bf16_t* wOutb  = (bf16_t*)alloc((size_t)HIDDEN * D_REC * 2);
    bf16_t* wMGb   = (bf16_t*)alloc((size_t)INTER * HIDDEN * 2);
    bf16_t* wMUb   = (bf16_t*)alloc((size_t)INTER * HIDDEN * 2);
    bf16_t* wMDb   = (bf16_t*)alloc((size_t)HIDDEN * INTER * 2);
    bf16_t* h_bf   = (bf16_t*)alloc((size_t)MTOK * HIDDEN * 2);   // reused for h2
    bf16_t* xin_bf = (bf16_t*)alloc((size_t)MTOK * D_REC * 2);
    bf16_t* gate_bf= (bf16_t*)alloc((size_t)MTOK * D_REC * 2);    // becomes y in-place
    float*  a_f32  = (float*)alloc((size_t)MTOK * D_REC * 4);     // 128 MiB
    bf16_t* mact   = (bf16_t*)a_f32;  // alias: MTOK*INTER*2 == MTOK*D_REC*4 bytes
    float*  x2     = (float*)alloc((size_t)MTOK * HIDDEN * 4);

    auto conv = [&](const float* s, bf16_t* d, int n) {
        convert_f32_bf16<<<(n + 255) / 256, 256, 0, stream>>>(s, d, n);
    };
    conv(w_in,   wInb,   D_REC * HIDDEN);
    conv(w_gate, wGateb, D_REC * HIDDEN);
    conv(w_a,    wAb,    D_REC * D_REC);
    conv(w_out,  wOutb,  HIDDEN * D_REC);
    conv(w_mg,   wMGb,   INTER * HIDDEN);
    conv(w_mu,   wMUb,   INTER * HIDDEN);
    conv(w_md,   wMDb,   HIDDEN * INTER);

    // h = rmsnorm(x, ln1_w)
    rmsnorm_kernel<<<MTOK, 256, 0, stream>>>(x, ln1_w, h_bf);

    // x_in = h @ w_in^T (bf16) ; gate = sigmoid(h @ w_gate^T) (bf16)
    {
        dim3 g(D_REC / 64, MTOK / 128);
        gemm_dual<<<g, 128, 0, stream>>>(h_bf, wInb, wGateb, D_REC, HIDDEN,
                                         xin_bf, gate_bf, 0);
    }

    // a = sigmoid(a_param + x_in @ w_a^T) (f32)
    {
        dim3 g(D_REC / 64, MTOK / 128);
        gemm_single<<<g, 128, 0, stream>>>(xin_bf, wAb, D_REC, D_REC,
                                           a_f32, a_param, nullptr, 0);
    }

    // y = gate * hawk_scan(x_in, a)  (in place over gate buffer)
    hawk_scan_kernel<<<(NB * D_REC) / 256, 256, 0, stream>>>(a_f32, xin_bf, gate_bf);

    // x2 = x + y @ w_out^T (f32)
    {
        dim3 g(HIDDEN / 64, MTOK / 128);
        gemm_single<<<g, 128, 0, stream>>>(gate_bf, wOutb, HIDDEN, D_REC,
                                           x2, nullptr, x, 1);
    }

    // h2 = rmsnorm(x2, ln2_w)
    rmsnorm_kernel<<<MTOK, 256, 0, stream>>>(x2, ln2_w, h_bf);

    // mact = silu(h2 @ w_mg^T) * (h2 @ w_mu^T)  (bf16, aliases a_f32 region)
    {
        dim3 g(INTER / 64, MTOK / 128);
        gemm_dual<<<g, 128, 0, stream>>>(h_bf, wMGb, wMUb, INTER, HIDDEN,
                                         mact, nullptr, 1);
    }

    // out = x2 + mact @ w_md^T (f32)
    {
        dim3 g(HIDDEN / 64, MTOK / 128);
        gemm_single<<<g, 128, 0, stream>>>(mact, wMDb, HIDDEN, INTER,
                                           (float*)d_out, nullptr, x2, 1);
    }
}
